// Aligner_47682726920868
// MI455X (gfx1250) — matrix operand order repaired and checked by reading, not compiled, not run on hardware
//
#include <hip/hip_runtime.h>
#include <hip/hip_bf16.h>
#include <math.h>
#include <stdint.h>

// ---------------- problem constants ----------------
#define BSZ   16
#define TPH   512
#define TMEL  2048
#define DIM   256
#define NMELS 80
#define KPAD  96          // 80 padded to multiple of 32 for K of WMMA
#define NEGV  (-1e9f)
#define LOGCONST (-73.515082656373f)   // -0.5*log(2*pi)*80

typedef __attribute__((ext_vector_type(16))) _Float16 v16h;
typedef __attribute__((ext_vector_type(8)))  _Float16 v8h;
typedef __attribute__((ext_vector_type(8)))  float    v8f;

// ---------------- WMMA helpers (wave32, 16x16x32 f16 -> f32) ----------------
// A: row-major [M x K] tile; lanes 0-15 M=lane K in {0..7,16..23},
// lanes 16-31 M=lane-16, K in {8..15,24..31}. Two 16B loads per lane.
__device__ __forceinline__ v16h load_A16x32(const _Float16* tile, int ld, int lane, int k0) {
    int m   = lane & 15;
    int khi = (lane >> 4) << 3;              // 0 or 8
    const _Float16* p = tile + (size_t)m * ld + k0 + khi;
    v8h lo = *(const v8h*)p;
    v8h hi = *(const v8h*)(p + 16);
    v16h r;
#pragma unroll
    for (int i = 0; i < 8; ++i) { r[i] = lo[i]; r[i + 8] = hi[i]; }
    return r;
}
// B sourced from row-major [N x K] matrix (we multiply by its transpose):
// lane's column n = lane&15 is a K-row; lanes 0-15 K in {0..7,16..23}, 16-31 K in {8..15,24..31}.
__device__ __forceinline__ v16h load_Bt16x32(const _Float16* tile, int ld, int lane, int k0) {
    int n  = lane & 15;
    int kb = (lane >> 4) << 3;               // 0 or 8
    const _Float16* p = tile + (size_t)n * ld + k0 + kb;
    v8h lo = *(const v8h*)p;
    v8h hi = *(const v8h*)(p + 16);
    v16h r;
#pragma unroll
    for (int i = 0; i < 8; ++i) { r[i] = lo[i]; r[i + 8] = hi[i]; }
    return r;
}
__device__ __forceinline__ v8f wmma_f16(v16h a, v16h b, v8f c) {
    return __builtin_amdgcn_wmma_f32_16x16x32_f16(false, a, false, b, (short)0, c, false, false);
}

// ---------------- small utility kernels ----------------
__global__ void k_zero_f32(float* __restrict__ p, size_t n) {
    size_t i = (size_t)blockIdx.x * blockDim.x + threadIdx.x;
    size_t stride = (size_t)gridDim.x * blockDim.x;
    for (; i < n; i += stride) p[i] = 0.f;
}
__global__ void k_cast_f16(const float* __restrict__ s, _Float16* __restrict__ d, int n) {
    int i = blockIdx.x * blockDim.x + threadIdx.x;
    if (i < n) d[i] = (_Float16)s[i];
}
__global__ void k_embed(const int* __restrict__ ph, const float* __restrict__ emb,
                        _Float16* __restrict__ x16) {
    int bt = blockIdx.x;                 // B*TPH
    int d  = threadIdx.x;                // DIM
    int id = ph[bt];
    x16[(size_t)bt * DIM + d] = (_Float16)emb[(size_t)id * DIM + d];
}
__global__ void k_melpad(const float* __restrict__ mel, _Float16* __restrict__ melp) {
    int bt = blockIdx.x;                 // B*TMEL
    int k  = threadIdx.x;                // KPAD
    melp[(size_t)bt * KPAD + k] = (_Float16)((k < NMELS) ? mel[(size_t)bt * NMELS + k] : 0.f);
}

// ---------------- xproj = x @ Wih^T + bih (both directions) ----------------
__global__ void k_xproj(const _Float16* __restrict__ x16,
                        const _Float16* __restrict__ Wf, const _Float16* __restrict__ Wb,
                        const float* __restrict__ bf, const float* __restrict__ bb,
                        float* __restrict__ outf, float* __restrict__ outb) {
    int wave = threadIdx.x >> 5, lane = threadIdx.x & 31;
    int tile = blockIdx.x * 4 + wave;            // 24576 tiles total
    int mt = tile / 48, nt = tile % 48;
    const _Float16* W    = blockIdx.z ? Wb   : Wf;
    const float*    bias = blockIdx.z ? bb   : bf;
    float*          out  = blockIdx.z ? outb : outf;
    const _Float16* At = x16 + (size_t)(mt * 16) * DIM;
    const _Float16* Bt = W   + (size_t)(nt * 16) * DIM;
    v8f acc = {};
#pragma unroll
    for (int k0 = 0; k0 < DIM; k0 += 32) {
        __builtin_prefetch(At + (size_t)(lane & 15) * DIM + k0 + 64, 0, 1);
        v16h a = load_A16x32(At, DIM, lane, k0);
        v16h b = load_Bt16x32(Bt, DIM, lane, k0);
        acc = wmma_f16(a, b, acc);
    }
    int n = nt * 16 + (lane & 15);
    float bv = bias[n];
    int mbase = mt * 16 + ((lane >> 4) << 3);
#pragma unroll
    for (int r = 0; r < 8; ++r)
        out[(size_t)(mbase + r) * (3 * DIM) + n] = acc[r] + bv;
}

// ---------------- GRU scan (one workgroup per direction, 16 waves) ----------------
// wave w owns hidden cols [16w,16w+16). Its 24 Whh B-fragments (3 gates x K=256/32)
// are loop-invariant, so they are preloaded into registers and held across the
// entire 512-step scan: per step only LDS A-loads + 24 WMMAs on the serial chain.
__global__ void k_gru(const _Float16* __restrict__ Whh16f, const _Float16* __restrict__ Whh16b,
                      const float* __restrict__ bhhf, const float* __restrict__ bhhb,
                      const float* __restrict__ xpf, const float* __restrict__ xpb,
                      _Float16* __restrict__ hcat) {
    __shared__ float    h32[BSZ * DIM];
    __shared__ _Float16 h16[BSZ * DIM];
    int dir = blockIdx.x;
    const _Float16* Whh = dir ? Whh16b : Whh16f;
    const float*    bhh = dir ? bhhb   : bhhf;
    const float*    xp  = dir ? xpb    : xpf;
    int wave = threadIdx.x >> 5, lane = threadIdx.x & 31;
    int c0 = wave * 16;
    for (int i = threadIdx.x; i < BSZ * DIM; i += blockDim.x) {
        h32[i] = 0.f; h16[i] = (_Float16)0.f;
    }
    // preload loop-invariant Whh fragments into registers (24 x v16h = 192 VGPRs)
    v16h bfr[3][8];
#pragma unroll
    for (int g = 0; g < 3; ++g)
#pragma unroll
        for (int kk = 0; kk < 8; ++kk)
            bfr[g][kk] = load_Bt16x32(Whh + (size_t)(g * DIM + c0) * DIM, DIM, lane, kk * 32);
    __syncthreads();
    int n = lane & 15;
    int mhalf = (lane >> 4) << 3;
    int col = c0 + n;
    float br_ = bhh[col], bz_ = bhh[DIM + col], bn_ = bhh[2 * DIM + col];
    for (int s = 0; s < TPH; ++s) {
        int t = dir ? (TPH - 1 - s) : s;
        v8f ar = {}, az = {}, an = {};
#pragma unroll
        for (int kk = 0; kk < 8; ++kk) {
            v16h a = load_A16x32((const _Float16*)h16, DIM, lane, kk * 32);
            ar = wmma_f16(a, bfr[0][kk], ar);
            az = wmma_f16(a, bfr[1][kk], az);
            an = wmma_f16(a, bfr[2][kk], an);
        }
        __syncthreads();   // all waves done reading h16 for this step
#pragma unroll
        for (int r = 0; r < 8; ++r) {
            int b = mhalf + r;
            const float* xrow = xp + ((size_t)b * TPH + t) * (3 * DIM);
            float xr = xrow[col], xz = xrow[DIM + col], xn = xrow[2 * DIM + col];
            float hr = ar[r] + br_, hz = az[r] + bz_, hn = an[r] + bn_;
            float rg = 1.f / (1.f + __expf(-(xr + hr)));
            float zg = 1.f / (1.f + __expf(-(xz + hz)));
            float ng = tanhf(xn + rg * hn);
            float hold = h32[b * DIM + col];
            float hnew = (1.f - zg) * ng + zg * hold;
            h32[b * DIM + col] = hnew;
            h16[b * DIM + col] = (_Float16)hnew;
            hcat[((size_t)b * TPH + t) * (2 * DIM) + dir * DIM + col] = (_Float16)hnew;
        }
        __syncthreads();   // h updated before next step's reads
    }
}

// ---------------- FC: [8192,512] @ Wfc^T[512->80] + bfc ----------------
__global__ void k_fc(const _Float16* __restrict__ hcat, const _Float16* __restrict__ Wfc16,
                     const float* __restrict__ bfc,
                     float* __restrict__ xfc32, _Float16* __restrict__ xfc16p) {
    int wave = threadIdx.x >> 5, lane = threadIdx.x & 31;
    int tile = blockIdx.x * 4 + wave;            // 2560 tiles
    int mt = tile / 5, nt = tile % 5;
    const _Float16* At = hcat  + (size_t)(mt * 16) * (2 * DIM);
    const _Float16* Bt = Wfc16 + (size_t)(nt * 16) * (2 * DIM);
    v8f acc = {};
#pragma unroll
    for (int k0 = 0; k0 < 2 * DIM; k0 += 32)
        acc = wmma_f16(load_A16x32(At, 2 * DIM, lane, k0),
                       load_Bt16x32(Bt, 2 * DIM, lane, k0), acc);
    int nn = nt * 16 + (lane & 15);              // < 80 always
    float bv = bfc[nn];
    int mbase = mt * 16 + ((lane >> 4) << 3);
#pragma unroll
    for (int r = 0; r < 8; ++r) {
        float v = acc[r] + bv;
        int row = mbase + r;
        xfc32[(size_t)row * NMELS + nn]  = v;
        xfc16p[(size_t)row * KPAD + nn] = (_Float16)v;   // pad cols pre-zeroed
    }
}

// ---------------- -0.5*row sums of squares ----------------
__global__ void k_rowsq(const float* __restrict__ src, float* __restrict__ out,
                        int rows, int len) {
    int i = blockIdx.x * blockDim.x + threadIdx.x;
    if (i >= rows) return;
    const float* p = src + (size_t)i * len;
    float s = 0.f;
    for (int k = 0; k < len; ++k) s += p[k] * p[k];
    out[i] = -0.5f * s;
}

// ---------------- log-prior GEMM; A-tile staged to LDS with async copies ----------------
// Block = 4 waves sharing one A-tile (mt) and covering 4 consecutive nt tiles.
// A tile (16x96 f16 = 3KB) is staged via global_load_async_to_lds_b128 (ASYNCcnt),
// then consumed via ds_load_b128; B streams straight from global.
// Writes value TRANSPOSED [B, T_mel, T_ph] so the DP scan reads contiguously.
__global__ void k_logprior(const _Float16* __restrict__ xfc16p, const _Float16* __restrict__ melp,
                           const float* __restrict__ msq, const float* __restrict__ ysq,
                           const unsigned char* __restrict__ ph_mask,
                           const unsigned char* __restrict__ mel_mask,
                           float* __restrict__ value_t) {
    __shared__ _Float16 Ash[16 * KPAD];          // 3KB
    int wave = threadIdx.x >> 5, lane = threadIdx.x & 31;
    int b  = blockIdx.y;
    int mt = blockIdx.x >> 5;                    // 0..31 (T_ph tiles)
    int nt = (blockIdx.x & 31) * 4 + wave;       // 0..127 (T_mel tiles)
    const _Float16* Ag = xfc16p + ((size_t)b * TPH + mt * 16) * KPAD;
    if (wave == 0) {
        unsigned long long ga = (unsigned long long)(uintptr_t)Ag;
        unsigned la0 = (unsigned)(uintptr_t)(&Ash[0]);
#pragma unroll
        for (int c = 0; c < 6; ++c) {            // 6 * (32 lanes * 16B) = 3072B
            unsigned off = (unsigned)(c * 512 + lane * 16);
            unsigned la = la0 + off;
            unsigned long long g = ga + off;
            asm volatile("global_load_async_to_lds_b128 %0, %1, off"
                         :: "v"(la), "v"(g) : "memory");
        }
        asm volatile("s_wait_asynccnt 0x0" ::: "memory");
    }
    __syncthreads();
    const _Float16* Bt = melp + ((size_t)b * TMEL + nt * 16) * KPAD;
    v8f acc = {};
#pragma unroll
    for (int k0 = 0; k0 < KPAD; k0 += 32)
        acc = wmma_f16(load_A16x32((const _Float16*)Ash, KPAD, lane, k0),
                       load_Bt16x32(Bt, KPAD, lane, k0), acc);
    int t = nt * 16 + (lane & 15);
    float ys = ysq[b * TMEL + t];
    unsigned char mm = mel_mask[b * TMEL + t];
    int pbase = mt * 16 + ((lane >> 4) << 3);
#pragma unroll
    for (int r = 0; r < 8; ++r) {
        int p = pbase + r;
        unsigned char pm = ph_mask[b * TPH + p];
        float v = acc[r] + ys + msq[b * TPH + p] + LOGCONST;
        v = (pm && mm) ? v : 0.f;                // attn_mask multiply
        if (!pm) v = NEGV;                       // maximum_path masking
        value_t[((size_t)b * TMEL + t) * TPH + p] = v;
    }
}

// ---------------- DP forward scan (per-batch block, 512 threads) ----------------
__global__ void k_dp(const float* __restrict__ value_t, unsigned char* __restrict__ diag8) {
    __shared__ float buf[2][TPH];
    int b = blockIdx.x, p = threadIdx.x;
    buf[0][p] = (p == 0) ? 0.f : NEGV;
    __syncthreads();
    int cur = 0;
    for (int t = 0; t < TMEL; ++t) {
        float c    = buf[cur][p];
        float left = (p == 0) ? NEGV : buf[cur][p - 1];
        float v    = value_t[((size_t)b * TMEL + t) * TPH + p];
        diag8[((size_t)b * TMEL + t) * TPH + p] = (unsigned char)(left > c);
        buf[cur ^ 1][p] = v + fmaxf(c, left);
        cur ^= 1;
        __syncthreads();
    }
}

// ---------------- backtrack: one thread per batch ----------------
__global__ void k_backtrack(const unsigned char* __restrict__ diag8,
                            const unsigned char* __restrict__ ph_mask,
                            const unsigned char* __restrict__ mel_mask,
                            int* __restrict__ idx_map, float* __restrict__ attn) {
    int b = threadIdx.x;
    if (b >= BSZ) return;
    int tx = 0, ty = 0;
    for (int i = 0; i < TPH;  ++i) tx += ph_mask[b * TPH + i]   ? 1 : 0;
    for (int i = 0; i < TMEL; ++i) ty += mel_mask[b * TMEL + i] ? 1 : 0;
    int idx = tx - 1;
    for (int t = TMEL - 1; t >= 0; --t) {
        if (t < ty) {
            idx_map[b * TMEL + t] = idx;
            attn[((size_t)b * TPH + idx) * TMEL + t] = 1.0f;
            idx -= (int)diag8[((size_t)b * TMEL + t) * TPH + idx];
        } else {
            idx_map[b * TMEL + t] = -1;
        }
    }
}

// ---------------- expand (gather) + masked MSE accumulation ----------------
__global__ void k_expand(const int* __restrict__ idx_map, const float* __restrict__ xfc32,
                         const float* __restrict__ mel,
                         float* __restrict__ expanded, float* __restrict__ rawloss) {
    __shared__ float red[NMELS];
    int bt = blockIdx.x;                 // B*TMEL
    int m  = threadIdx.x;                // NMELS
    int p  = idx_map[bt];
    int b  = bt / TMEL;
    float e = 0.f, d2 = 0.f;
    if (p >= 0) {
        e = xfc32[((size_t)b * TPH + p) * NMELS + m];
        float d = e - mel[(size_t)bt * NMELS + m];
        d2 = d * d;
    }
    expanded[(size_t)bt * NMELS + m] = e;
    red[m] = d2;
    __syncthreads();
    if (m == 0) {
        float s = 0.f;
        for (int i = 0; i < NMELS; ++i) s += red[i];
        atomicAdd(rawloss, s);
    }
}

__global__ void k_masksum(const unsigned char* __restrict__ mel_mask, unsigned int* __restrict__ cnt) {
    int i = blockIdx.x * blockDim.x + threadIdx.x;
    if (i < BSZ * TMEL && mel_mask[i]) atomicAdd(cnt, 1u);
}
__global__ void k_finalize(const float* __restrict__ rawloss, const unsigned int* __restrict__ cnt,
                           float* __restrict__ out_loss) {
    if (threadIdx.x == 0) out_loss[0] = rawloss[0] / ((float)cnt[0] * (float)NMELS);
}

// ---------------- workspace layout (bytes) ----------------
#define ALIGN256(x) (((x) + 255) & ~(size_t)255)
static const size_t OFF_X16    = 0;                                                   // 8192*256 f16
static const size_t OFF_WIHF   = ALIGN256(OFF_X16    + (size_t)BSZ*TPH*DIM*2);        // 768*256 f16
static const size_t OFF_WIHB   = ALIGN256(OFF_WIHF   + (size_t)3*DIM*DIM*2);
static const size_t OFF_WHHF   = ALIGN256(OFF_WIHB   + (size_t)3*DIM*DIM*2);
static const size_t OFF_WHHB   = ALIGN256(OFF_WHHF   + (size_t)3*DIM*DIM*2);
static const size_t OFF_WFC    = ALIGN256(OFF_WHHB   + (size_t)3*DIM*DIM*2);          // 80*512 f16
static const size_t OFF_XPF    = ALIGN256(OFF_WFC    + (size_t)NMELS*2*DIM*2);        // 8192*768 f32
static const size_t OFF_XPB    = ALIGN256(OFF_XPF    + (size_t)BSZ*TPH*3*DIM*4);
static const size_t OFF_HCAT   = ALIGN256(OFF_XPB    + (size_t)BSZ*TPH*3*DIM*4);      // 8192*512 f16
static const size_t OFF_XFC32  = ALIGN256(OFF_HCAT   + (size_t)BSZ*TPH*2*DIM*2);      // 8192*80 f32
static const size_t OFF_XFC16P = ALIGN256(OFF_XFC32  + (size_t)BSZ*TPH*NMELS*4);      // 8192*96 f16
static const size_t OFF_MELP   = ALIGN256(OFF_XFC16P + (size_t)BSZ*TPH*KPAD*2);       // 32768*96 f16
static const size_t OFF_YSQ    = ALIGN256(OFF_MELP   + (size_t)BSZ*TMEL*KPAD*2);      // 32768 f32
static const size_t OFF_MSQ    = ALIGN256(OFF_YSQ    + (size_t)BSZ*TMEL*4);           // 8192 f32
static const size_t OFF_VALT   = ALIGN256(OFF_MSQ    + (size_t)BSZ*TPH*4);            // 16M f32
static const size_t OFF_DIAG   = ALIGN256(OFF_VALT   + (size_t)BSZ*TMEL*TPH*4);       // 16M u8
static const size_t OFF_IDX    = ALIGN256(OFF_DIAG   + (size_t)BSZ*TMEL*TPH);         // 32768 i32
static const size_t OFF_SCAL   = ALIGN256(OFF_IDX    + (size_t)BSZ*TMEL*4);           // rawloss, cnt

extern "C" void kernel_launch(void* const* d_in, const int* in_sizes, int n_in,
                              void* d_out, int out_size, void* d_ws, size_t ws_size,
                              hipStream_t stream) {
    (void)in_sizes; (void)n_in; (void)out_size; (void)ws_size;
    const int*   ph       = (const int*)d_in[0];
    const float* mel      = (const float*)d_in[1];
    const unsigned char* ph_mask  = (const unsigned char*)d_in[2];
    const unsigned char* mel_mask = (const unsigned char*)d_in[3];
    const float* emb   = (const float*)d_in[4];
    const float* Wih_f = (const float*)d_in[5];
    const float* Whh_f = (const float*)d_in[6];
    const float* bih_f = (const float*)d_in[7];
    const float* bhh_f = (const float*)d_in[8];
    const float* Wih_b = (const float*)d_in[9];
    const float* Whh_b = (const float*)d_in[10];
    const float* bih_b = (const float*)d_in[11];
    const float* bhh_b = (const float*)d_in[12];
    const float* Wfc   = (const float*)d_in[13];
    const float* bfc   = (const float*)d_in[14];

    char* ws = (char*)d_ws;
    _Float16* x16    = (_Float16*)(ws + OFF_X16);
    _Float16* wihf16 = (_Float16*)(ws + OFF_WIHF);
    _Float16* wihb16 = (_Float16*)(ws + OFF_WIHB);
    _Float16* whhf16 = (_Float16*)(ws + OFF_WHHF);
    _Float16* whhb16 = (_Float16*)(ws + OFF_WHHB);
    _Float16* wfc16  = (_Float16*)(ws + OFF_WFC);
    float*    xpf    = (float*)   (ws + OFF_XPF);
    float*    xpb    = (float*)   (ws + OFF_XPB);
    _Float16* hcat   = (_Float16*)(ws + OFF_HCAT);
    float*    xfc32  = (float*)   (ws + OFF_XFC32);
    _Float16* xfc16p = (_Float16*)(ws + OFF_XFC16P);
    _Float16* melp   = (_Float16*)(ws + OFF_MELP);
    float*    ysq    = (float*)   (ws + OFF_YSQ);
    float*    msq    = (float*)   (ws + OFF_MSQ);
    float*    valt   = (float*)   (ws + OFF_VALT);
    unsigned char* diag8 = (unsigned char*)(ws + OFF_DIAG);
    int*      idxmap = (int*)     (ws + OFF_IDX);
    float*    rawloss = (float*)  (ws + OFF_SCAL);
    unsigned int* cnt = (unsigned int*)(ws + OFF_SCAL + 4);

    float* attn     = (float*)d_out;
    float* expanded = attn + (size_t)BSZ * TPH * TMEL;
    float* out_loss = expanded + (size_t)BSZ * TMEL * NMELS;

    // zero attn region of d_out, the f16 pad buffer, and scalar accumulators
    k_zero_f32<<<2048, 256, 0, stream>>>(attn, (size_t)BSZ * TPH * TMEL);
    k_zero_f32<<<256, 256, 0, stream>>>((float*)xfc16p, (size_t)BSZ * TPH * KPAD / 2);
    k_zero_f32<<<1, 32, 0, stream>>>((float*)(ws + OFF_SCAL), 2);

    // cast weights to f16
    int nw = 3 * DIM * DIM;
    k_cast_f16<<<(nw + 255) / 256, 256, 0, stream>>>(Wih_f, wihf16, nw);
    k_cast_f16<<<(nw + 255) / 256, 256, 0, stream>>>(Wih_b, wihb16, nw);
    k_cast_f16<<<(nw + 255) / 256, 256, 0, stream>>>(Whh_f, whhf16, nw);
    k_cast_f16<<<(nw + 255) / 256, 256, 0, stream>>>(Whh_b, whhb16, nw);
    int nf = NMELS * 2 * DIM;
    k_cast_f16<<<(nf + 255) / 256, 256, 0, stream>>>(Wfc, wfc16, nf);

    // embedding gather + mel pad/cast
    k_embed<<<BSZ * TPH, DIM, 0, stream>>>(ph, emb, x16);
    k_melpad<<<BSZ * TMEL, KPAD, 0, stream>>>(mel, melp);

    // xproj for both directions: 512*48 tiles / 4 waves per block
    dim3 gx(512 * 48 / 4, 1, 2);
    k_xproj<<<gx, 128, 0, stream>>>(x16, wihf16, wihb16, bih_f, bih_b, xpf, xpb);

    // GRU scan: one workgroup per direction
    k_gru<<<2, 512, 0, stream>>>(whhf16, whhb16, bhh_f, bhh_b, xpf, xpb, hcat);

    // FC: 512*5 tiles / 4 waves per block
    k_fc<<<512 * 5 / 4, 128, 0, stream>>>(hcat, wfc16, bfc, xfc32, xfc16p);

    // -0.5*||.||^2 rows
    k_rowsq<<<(BSZ * TMEL + 255) / 256, 256, 0, stream>>>(mel, ysq, BSZ * TMEL, NMELS);
    k_rowsq<<<(BSZ * TPH + 255) / 256, 256, 0, stream>>>(xfc32, msq, BSZ * TPH, NMELS);

    // log-prior GEMM -> value_t [B, T_mel, T_ph]; 32 mt-tiles x 32 nt-groups per batch
    dim3 gl(32 * 32, BSZ, 1);
    k_logprior<<<gl, 128, 0, stream>>>(xfc16p, melp, msq, ysq, ph_mask, mel_mask, valt);

    // DP forward + backtrack
    k_dp<<<BSZ, TPH, 0, stream>>>(valt, diag8);
    k_backtrack<<<1, 32, 0, stream>>>(diag8, ph_mask, mel_mask, idxmap, attn);

    // expand + loss
    k_masksum<<<(BSZ * TMEL + 255) / 256, 256, 0, stream>>>(mel_mask, cnt);
    k_expand<<<BSZ * TMEL, NMELS, 0, stream>>>(idxmap, xfc32, mel, expanded, rawloss);
    k_finalize<<<1, 32, 0, stream>>>(rawloss, cnt, out_loss);
}
